// MoeBlock_24816321036510
// MI455X (gfx1250) — matrix operand order repaired and checked by reading, not compiled, not run on hardware
//
#include <hip/hip_runtime.h>
#include <hip/hip_bf16.h>
#include <math.h>

#define T_TOK 4096
#define H_DIM 1024
#define F_DIM 4096
#define NEXP  8
#define TOPK  2
#define CAPTOT 8320            // 8192 assignments + up to 8*16 padding
#define NTILES (CAPTOT/16)     // 520

typedef __attribute__((ext_vector_type(16))) __bf16 v16bf;
typedef __attribute__((ext_vector_type(8)))  __bf16 v8bf;
typedef __attribute__((ext_vector_type(8)))  float  v8f;

__device__ inline v8f wmma_bf16(v16bf a, v16bf b, v8f c) {
  // D = A(16x32 bf16) * B(32x16 bf16) + C(16x16 f32)
  return __builtin_amdgcn_wmma_f32_16x16x32_bf16(
      /*neg_a=*/false, a, /*neg_b=*/false, b,
      /*c_mod=*/(short)0, c, /*reuse_a=*/false, /*reuse_b=*/false);
}

// Per-lane A fragment: 16 contiguous k (k = 16*half + s), matching B packing.
__device__ inline v16bf loadA16(const __bf16* row, int kk, int half) {
  v8bf a0 = *(const v8bf*)(row + kk + 16 * half);
  v8bf a1 = *(const v8bf*)(row + kk + 16 * half + 8);
  return __builtin_shufflevector(a0, a1, 0, 1, 2, 3, 4, 5, 6, 7,
                                 8, 9, 10, 11, 12, 13, 14, 15);
}

// ---------------- weight pre-conversion (fp32 -> bf16 hi/lo, B-fragment layout) ----
// dst layout: [E][K/32][N][32] bf16 ; a lane's B fragment = 16 contiguous bf16
// at (((e*ktiles + kt)*N + n)*32 + 16*half).

__global__ void k_convert(const float* __restrict__ src, __bf16* __restrict__ dh,
                          __bf16* __restrict__ dl, int ktiles, int ncols) {
  size_t u = (size_t)blockIdx.x * 256 + threadIdx.x;
  size_t total = (size_t)NEXP * ktiles * ncols;
  if (u >= total) return;
  int n = (int)(u % ncols);
  size_t t2 = u / ncols;
  int kt = (int)(t2 % ktiles);
  int e = (int)(t2 / ktiles);
  const float* s = src + (((size_t)e * ktiles + kt) * 32) * ncols + n;
  __bf16 hb[32], lb[32];
#pragma unroll
  for (int k2 = 0; k2 < 32; ++k2) {
    float f = s[(size_t)k2 * ncols];
    __bf16 h = (__bf16)f;
    hb[k2] = h;
    lb[k2] = (__bf16)(f - (float)h);
  }
  __bf16* oh = dh + u * 32;
  __bf16* ol = dl + u * 32;
#pragma unroll
  for (int q = 0; q < 4; ++q) {
    *(v8bf*)(oh + q * 8) = *(v8bf*)(hb + q * 8);
    *(v8bf*)(ol + q * 8) = *(v8bf*)(lb + q * 8);
  }
}

// ---------------- routing ----------------

__global__ void k_zero(int* cnt) {
  if (threadIdx.x < NEXP) cnt[threadIdx.x] = 0;
}

__global__ void k_router(const float* __restrict__ x, const float* __restrict__ wg,
                         float* __restrict__ logits_out, int* __restrict__ tIdx,
                         float* __restrict__ tW, int* __restrict__ cnt) {
  int wave = threadIdx.x >> 5, lane = threadIdx.x & 31;
  int t = blockIdx.x * 4 + wave;
  if (t >= T_TOK) return;
  float acc[NEXP];
#pragma unroll
  for (int e = 0; e < NEXP; ++e) acc[e] = 0.f;
  for (int h = lane; h < H_DIM; h += 32) {
    float xv = x[(size_t)t * H_DIM + h];
#pragma unroll
    for (int e = 0; e < NEXP; ++e) acc[e] += xv * wg[h * NEXP + e];
  }
#pragma unroll
  for (int e = 0; e < NEXP; ++e)
    for (int off = 16; off > 0; off >>= 1) acc[e] += __shfl_down(acc[e], off, 32);
  if (lane == 0) {
    float mx = acc[0];
#pragma unroll
    for (int e = 1; e < NEXP; ++e) mx = fmaxf(mx, acc[e]);
    float p[NEXP], sum = 0.f;
#pragma unroll
    for (int e = 0; e < NEXP; ++e) { p[e] = expf(acc[e] - mx); sum += p[e]; }
#pragma unroll
    for (int e = 0; e < NEXP; ++e) {
      p[e] /= sum;
      logits_out[(size_t)t * NEXP + e] = acc[e];
    }
    int i0 = 0;
#pragma unroll
    for (int e = 1; e < NEXP; ++e) if (p[e] > p[i0]) i0 = e;
    int i1 = (i0 == 0) ? 1 : 0;
#pragma unroll
    for (int e = 0; e < NEXP; ++e) if (e != i0 && p[e] > p[i1]) i1 = e;
    float s2 = p[i0] + p[i1];
    tIdx[t * 2 + 0] = i0; tW[t * 2 + 0] = p[i0] / s2;
    tIdx[t * 2 + 1] = i1; tW[t * 2 + 1] = p[i1] / s2;
    atomicAdd(&cnt[i0], 1);
    atomicAdd(&cnt[i1], 1);
  }
}

__global__ void k_offsets(const int* __restrict__ cnt, int* __restrict__ offs,
                          int* __restrict__ cnt2) {
  if (threadIdx.x == 0) {
    int off = 0;
    for (int e = 0; e < NEXP; ++e) { offs[e] = off; off += ((cnt[e] + 15) >> 4) << 4; }
    offs[NEXP] = off;
  }
  if (threadIdx.x < NEXP) cnt2[threadIdx.x] = 0;
}

__global__ void k_fillpad(int* __restrict__ slotTok, float* __restrict__ slotW) {
  int i = blockIdx.x * 256 + threadIdx.x;
  if (i < CAPTOT) { slotTok[i] = -1; slotW[i] = 0.f; }
}

__global__ void k_scatter(const int* __restrict__ tIdx, const float* __restrict__ tW,
                          const int* __restrict__ offs, int* __restrict__ cnt2,
                          int* __restrict__ slotTok, float* __restrict__ slotW,
                          int* __restrict__ slotOf) {
  int t = blockIdx.x * 256 + threadIdx.x;
  if (t >= T_TOK) return;
#pragma unroll
  for (int k = 0; k < TOPK; ++k) {
    int e = tIdx[t * 2 + k];
    int pos = atomicAdd(&cnt2[e], 1);
    int s = offs[e] + pos;
    slotTok[s] = t; slotW[s] = tW[t * 2 + k]; slotOf[t * 2 + k] = s;
  }
}

// ---------------- fused GeGLU expert kernel (WMMA) ----------------
// 1 block = 16 gathered tokens of one expert, 4 waves.
// Weights arrive pre-split (bf16 hi/lo) in fragment order: one v16bf load per
// B fragment. 3x-bf16 split emulation: hi*hi + hi*lo + lo*hi.

__global__ void __launch_bounds__(128) k_moe(
    const float* __restrict__ x,
    const __bf16* __restrict__ wgh, const __bf16* __restrict__ wgl,
    const __bf16* __restrict__ wvh, const __bf16* __restrict__ wvl,
    const __bf16* __restrict__ w1h, const __bf16* __restrict__ w1l,
    const int* __restrict__ offs, const int* __restrict__ slotTok,
    const float* __restrict__ slotW, float* __restrict__ ybuf) {
  __shared__ __bf16 Xh[16][H_DIM];
  __shared__ __bf16 Xl[16][H_DIM];
  __shared__ __bf16 Ah[16][64];
  __shared__ __bf16 Al[16][64];
  __shared__ int   st[16];
  __shared__ float wsl[16];

  int g = blockIdx.x;
  int total = offs[NEXP];
  if (g * 16 >= total) return;            // uniform per block: WMMA EXEC stays all-1s
  int e = 0;
#pragma unroll
  for (int i = 1; i < NEXP; ++i) if (g * 16 >= offs[i]) e = i;

  int tid = threadIdx.x;
  int wv = tid >> 5, lane = tid & 31, half = lane >> 4, lcol = lane & 15;

  if (tid < 16) {
    int tok = slotTok[g * 16 + tid];
    st[tid] = tok;
    wsl[tid] = (tok >= 0) ? slotW[g * 16 + tid] : 0.f;
  }
  __syncthreads();
  // gather X tile -> LDS as bf16 hi/lo (split precision), once per block
  for (int i = tid; i < 16 * H_DIM; i += 128) {
    int r = i >> 10, c = i & (H_DIM - 1);
    int tok = st[r];
    float f = (tok >= 0) ? x[(size_t)tok * H_DIM + c] : 0.f;
    __bf16 h = (__bf16)f;
    Xh[r][c] = h;
    Xl[r][c] = (__bf16)(f - (float)h);
  }
  __syncthreads();

  const size_t upBase = (size_t)e * (H_DIM / 32);   // ktile base for wg/wv
  const size_t dnBase = (size_t)e * (F_DIM / 32);   // ktile base for w1

  v8f zero = {};
  v8f accd[16];
#pragma unroll
  for (int n = 0; n < 16; ++n) accd[n] = zero;

  for (int f0 = 0; f0 < F_DIM; f0 += 64) {
    // warm L2 for this chunk's down-proj weights while the up-GEMM runs
    {
      const size_t pfoff = ((dnBase + (f0 >> 5)) * H_DIM) * 32 + (size_t)tid * 512;
      __builtin_prefetch(w1h + pfoff, 0, 1);
      __builtin_prefetch(w1l + pfoff, 0, 1);
    }
    // ---- up-projections g and v: this wave's 16 columns ----
    v8f ag = {}; v8f av = {};
    int ncol = f0 + wv * 16 + lcol;
    for (int kk = 0; kk < H_DIM; kk += 32) {
      v16bf ah = loadA16(&Xh[lcol][0], kk, half);
      v16bf al = loadA16(&Xl[lcol][0], kk, half);
      size_t boff = (((upBase + (kk >> 5)) * F_DIM + ncol) << 5) + 16 * half;
      v16bf bgh = *(const v16bf*)(wgh + boff);
      v16bf bgl = *(const v16bf*)(wgl + boff);
      v16bf bvh = *(const v16bf*)(wvh + boff);
      v16bf bvl = *(const v16bf*)(wvl + boff);
      ag = wmma_bf16(ah, bgh, ag);
      ag = wmma_bf16(ah, bgl, ag);
      ag = wmma_bf16(al, bgh, ag);
      av = wmma_bf16(ah, bvh, av);
      av = wmma_bf16(ah, bvl, av);
      av = wmma_bf16(al, bvh, av);
    }
    // ---- act = gelu_exact(g) * v, stage chunk in LDS (bf16 hi/lo) ----
#pragma unroll
    for (int q = 0; q < 8; ++q) {
      float gg = ag[q], vv = av[q];
      float a = 0.5f * gg * (1.f + erff(gg * 0.70710678118654752f)) * vv;
      __bf16 h = (__bf16)a;
      int m = q + 8 * half, c = wv * 16 + lcol;          // C layout: M=q+8*half, N=lcol
      Ah[m][c] = h;
      Al[m][c] = (__bf16)(a - (float)h);
    }
    __syncthreads();
    // ---- down-projection: wave owns H cols [wv*256, wv*256+256) ----
#pragma unroll
    for (int nt = 0; nt < 16; ++nt) {
      int n0 = wv * 256 + nt * 16;
#pragma unroll
      for (int ks = 0; ks < 64; ks += 32) {
        v16bf ah = loadA16(&Ah[lcol][0], ks, half);
        v16bf al = loadA16(&Al[lcol][0], ks, half);
        size_t boff = (((dnBase + ((f0 + ks) >> 5)) * H_DIM + n0 + lcol) << 5) + 16 * half;
        v16bf bh = *(const v16bf*)(w1h + boff);
        v16bf bl = *(const v16bf*)(w1l + boff);
        accd[nt] = wmma_bf16(ah, bh, accd[nt]);
        accd[nt] = wmma_bf16(ah, bl, accd[nt]);
        accd[nt] = wmma_bf16(al, bh, accd[nt]);
      }
    }
    __syncthreads();
  }
  // ---- write per-assignment result, pre-scaled by routing weight ----
#pragma unroll
  for (int nt = 0; nt < 16; ++nt) {
#pragma unroll
    for (int q = 0; q < 8; ++q) {
      int m = q + 8 * half;
      int hcol = wv * 256 + nt * 16 + lcol;
      ybuf[(size_t)(g * 16 + m) * H_DIM + hcol] = accd[nt][q] * wsl[m];
    }
  }
}

__global__ void k_combine(const float* __restrict__ ybuf, const int* __restrict__ slotOf,
                          float* __restrict__ out) {
  int idx = blockIdx.x * 256 + threadIdx.x;
  if (idx >= T_TOK * H_DIM) return;
  int t = idx >> 10, h = idx & (H_DIM - 1);
  int s0 = slotOf[t * 2], s1 = slotOf[t * 2 + 1];
  out[idx] = ybuf[(size_t)s0 * H_DIM + h] + ybuf[(size_t)s1 * H_DIM + h];
}

// ---------------- launch ----------------

extern "C" void kernel_launch(void* const* d_in, const int* in_sizes, int n_in,
                              void* d_out, int out_size, void* d_ws, size_t ws_size,
                              hipStream_t stream) {
  const float* x   = (const float*)d_in[0];   // [T,H]
  const float* wg  = (const float*)d_in[1];   // [H,E]
  const float* w_g = (const float*)d_in[2];   // [E,H,F]
  const float* w_v = (const float*)d_in[3];   // [E,H,F]
  const float* w1  = (const float*)d_in[4];   // [E,F,H]
  float* out    = (float*)d_out;                      // [T,H]
  float* logits = out + (size_t)T_TOK * H_DIM;        // [T,E]

  char* p = (char*)d_ws;
  auto alloc = [&](size_t bytes) -> void* {
    void* r = (void*)p;
    p += (bytes + 255) & ~(size_t)255;
    return r;
  };
  const size_t WELEM = (size_t)NEXP * H_DIM * F_DIM;  // 33.5M elements per tensor

  int*    cnt     = (int*)alloc(8 * sizeof(int));
  int*    cnt2    = (int*)alloc(8 * sizeof(int));
  int*    offs    = (int*)alloc(16 * sizeof(int));
  int*    tIdx    = (int*)alloc((size_t)T_TOK * 2 * sizeof(int));
  float*  tW      = (float*)alloc((size_t)T_TOK * 2 * sizeof(float));
  int*    slotTok = (int*)alloc((size_t)CAPTOT * sizeof(int));
  float*  slotW   = (float*)alloc((size_t)CAPTOT * sizeof(float));
  int*    slotOf  = (int*)alloc((size_t)T_TOK * 2 * sizeof(int));
  float*  ybuf    = (float*)alloc((size_t)CAPTOT * H_DIM * sizeof(float)); // 34 MB
  __bf16* wgh     = (__bf16*)alloc(WELEM * sizeof(__bf16));  // 64 MB each
  __bf16* wgl     = (__bf16*)alloc(WELEM * sizeof(__bf16));
  __bf16* wvh     = (__bf16*)alloc(WELEM * sizeof(__bf16));
  __bf16* wvl     = (__bf16*)alloc(WELEM * sizeof(__bf16));
  __bf16* w1h     = (__bf16*)alloc(WELEM * sizeof(__bf16));
  __bf16* w1l     = (__bf16*)alloc(WELEM * sizeof(__bf16));

  const int cgrid = (int)((WELEM / 32 + 255) / 256);  // threads = E*(K/32)*N
  k_convert<<<cgrid, 256, 0, stream>>>(w_g, wgh, wgl, H_DIM / 32, F_DIM);
  k_convert<<<cgrid, 256, 0, stream>>>(w_v, wvh, wvl, H_DIM / 32, F_DIM);
  k_convert<<<cgrid, 256, 0, stream>>>(w1,  w1h, w1l, F_DIM / 32, H_DIM);

  k_zero   <<<1, 32, 0, stream>>>(cnt);
  k_router <<<T_TOK / 4, 128, 0, stream>>>(x, wg, logits, tIdx, tW, cnt);
  k_offsets<<<1, 32, 0, stream>>>(cnt, offs, cnt2);
  k_fillpad<<<(CAPTOT + 255) / 256, 256, 0, stream>>>(slotTok, slotW);
  k_scatter<<<(T_TOK + 255) / 256, 256, 0, stream>>>(tIdx, tW, offs, cnt2,
                                                     slotTok, slotW, slotOf);
  k_moe    <<<NTILES, 128, 0, stream>>>(x, wgh, wgl, wvh, wvl, w1h, w1l,
                                        offs, slotTok, slotW, ybuf);
  k_combine<<<(T_TOK * H_DIM) / 256, 256, 0, stream>>>(ybuf, slotOf, out);
}
